// TransformerBlock_44504451121348
// MI455X (gfx1250) — compile-verified, matrix operand order checked by reading, not run on hardware
//
#include <hip/hip_runtime.h>
#include <hip/hip_bf16.h>
#include <math.h>

// ---------------------------------------------------------------------------
// Types for CDNA5 WMMA (wave32): V_WMMA_F32_16X16X32_BF16 + TDM descriptors
// ---------------------------------------------------------------------------
typedef __bf16 bf16;
typedef __attribute__((ext_vector_type(16))) __bf16 v16bf;
typedef __attribute__((ext_vector_type(8)))  float  v8f;
typedef __attribute__((ext_vector_type(4)))  unsigned int v4u;
typedef __attribute__((ext_vector_type(4)))  unsigned int u32x4;
typedef __attribute__((ext_vector_type(8)))  int i32x8;
typedef __attribute__((ext_vector_type(4)))  int i32x4;

union FragU { v4u u[2]; v16bf v; };

// Problem constants
#define DIMD 1024
#define DIMH 4096
#define SEQ  4096
#define NB   4
#define MTOT (NB * SEQ)   // 16384

// GEMM tiling
#define BM 64
#define BN 128
#define BK 32
#define LDSS 40            // 32 + 8 pad bf16 elems -> 80B row stride (16B aligned, b128 conflict free)

// ---------------------------------------------------------------------------
// TDM: issue a 2-D tensor_load_to_lds (6-arg builtin on this toolchain).
//   Tile: tileRows rows x BK(=32) bf16 columns, row stride in memory = strideElems.
//   LDS: rows written contiguously (64B each) with 16B padding appended after
//        every 64B (pad_interval code 3, pad_amount code 3) -> 40-elem row stride.
// D# packing per CDNA5 ISA ch.8 (groups 0/1; groups 2/3 zero for 2-D).
// ---------------------------------------------------------------------------
__device__ __forceinline__ void tdm_load_tile_2d(unsigned lds_addr, const bf16* gptr,
                                                 unsigned tileRows,
                                                 unsigned tensorK, unsigned tensorRows,
                                                 unsigned strideElems) {
    unsigned long long ga = (unsigned long long)(uintptr_t)gptr;
    u32x4 g0;
    g0[0] = 0x1u;                                       // count=1 (valid descriptor)
    g0[1] = lds_addr;                                   // lds_addr [63:32]
    g0[2] = (unsigned)ga;                               // global_addr lo
    g0[3] = (unsigned)((ga >> 32) & 0x01FFFFFFu)        // global_addr [56:32]
            | (2u << 30);                               // type=2 ("image")
    i32x8 g1;
    g1[0] = (1 << 16)      // data_size = 1 -> 2 bytes (bf16)
          | (1 << 20)      // pad_enable
          | (3 << 22)      // pad_interval code 3 -> every 16 DWORDs (64B)
          | (3 << 25);     // pad_amount  code 3 -> 4 DWORDs (16B)
    g1[1] = (int)((tensorK & 0xFFFFu) << 16);                        // tensor_dim0 lo16 @bit48
    g1[2] = (int)((tensorK >> 16) | ((tensorRows & 0xFFFFu) << 16)); // dim0 hi / dim1 lo
    g1[3] = (int)((tensorRows >> 16) | ((unsigned)BK << 16));        // dim1 hi / tile_dim0=32
    g1[4] = (int)(tileRows & 0xFFFFu);                               // tile_dim1 (tile_dim2=0)
    g1[5] = (int)strideElems;                                        // tensor_dim0_stride lo32
    g1[6] = 0;                                                       // stride0 hi / stride1 lo
    g1[7] = 0;
    const i32x4 z4 = {0, 0, 0, 0};
    const i32x8 z8 = {0, 0, 0, 0, 0, 0, 0, 0};
    __builtin_amdgcn_tensor_load_to_lds(g0, g1, z4, z4, z8, 0);
}

// ---------------------------------------------------------------------------
// Weight pack: W fp32 [K][N] row-major  ->  Wt bf16 [N][K] row-major
// ---------------------------------------------------------------------------
__global__ __launch_bounds__(256) void pack_wT_kernel(const float* __restrict__ W,
                                                      bf16* __restrict__ Wt,
                                                      int K, int N) {
    __shared__ float tile[32][33];
    const int tx = threadIdx.x;          // 0..31
    const int ty = threadIdx.y;          // 0..7
    const int n0 = blockIdx.x * 32;
    const int k0 = blockIdx.y * 32;
#pragma unroll
    for (int i = 0; i < 32; i += 8)
        tile[ty + i][tx] = W[(size_t)(k0 + ty + i) * N + (n0 + tx)];
    __syncthreads();
#pragma unroll
    for (int i = 0; i < 32; i += 8)
        Wt[(size_t)(n0 + ty + i) * K + (k0 + tx)] = (bf16)tile[tx][ty + i];
}

// ---------------------------------------------------------------------------
// LayerNorm (fp32 in) -> bf16 out.  One block per row of D elements.
// ---------------------------------------------------------------------------
__global__ __launch_bounds__(256) void layernorm_bf16_kernel(const float* __restrict__ X,
                                                             const float* __restrict__ g,
                                                             const float* __restrict__ beta,
                                                             bf16* __restrict__ Y, int D) {
    const int row = blockIdx.x;
    const int tid = threadIdx.x;
    const float* x = X + (size_t)row * D;
    __shared__ float red[256];

    float s = 0.f;
    for (int c = tid; c < D; c += 256) s += x[c];
    red[tid] = s; __syncthreads();
    for (int o = 128; o > 0; o >>= 1) { if (tid < o) red[tid] += red[tid + o]; __syncthreads(); }
    const float mu = red[0] / (float)D;
    __syncthreads();

    float s2 = 0.f;
    for (int c = tid; c < D; c += 256) { float dd = x[c] - mu; s2 += dd * dd; }
    red[tid] = s2; __syncthreads();
    for (int o = 128; o > 0; o >>= 1) { if (tid < o) red[tid] += red[tid + o]; __syncthreads(); }
    const float rs = rsqrtf(red[0] / (float)D + 1e-5f);

    for (int c = tid; c < D; c += 256)
        Y[(size_t)row * D + c] = (bf16)((x[c] - mu) * rs * g[c] + beta[c]);
}

// ---------------------------------------------------------------------------
// Row softmax: fp32 scores [rows][cols] -> bf16 probs. One block per row.
// ---------------------------------------------------------------------------
__global__ __launch_bounds__(256) void softmax_bf16_kernel(const float* __restrict__ S,
                                                           bf16* __restrict__ P, int cols) {
    const int row = blockIdx.x;
    const int tid = threadIdx.x;
    const float* s = S + (size_t)row * cols;
    __shared__ float red[256];

    float m = -3.4e38f;
    for (int c = tid; c < cols; c += 256) m = fmaxf(m, s[c]);
    red[tid] = m; __syncthreads();
    for (int o = 128; o > 0; o >>= 1) { if (tid < o) red[tid] = fmaxf(red[tid], red[tid + o]); __syncthreads(); }
    const float rmax = red[0];
    __syncthreads();

    float sum = 0.f;
    for (int c = tid; c < cols; c += 256) sum += __expf(s[c] - rmax);
    red[tid] = sum; __syncthreads();
    for (int o = 128; o > 0; o >>= 1) { if (tid < o) red[tid] += red[tid + o]; __syncthreads(); }
    const float inv = 1.f / red[0];

    for (int c = tid; c < cols; c += 256)
        P[(size_t)row * cols + c] = (bf16)(__expf(s[c] - rmax) * inv);
}

// ---------------------------------------------------------------------------
// WMMA bf16 GEMM with TDM-fed, double-buffered LDS tiles:
//   C[M][N] = alpha * A[M][K] * Bt[N][K]^T  (+bias)(+GELU)(+residual)
// Block: 128 threads = 4 waves (2x2), tile 64x128, K-step 32.
// Wave 0 issues two tensor_load_to_lds per K-step (A tile 64x32, B tile 128x32)
// into the alternate buffer, then s_wait_tensorcnt(2) retires the current one.
// Each wave: 32x64 -> 8 x v_wmma_f32_16x16x32_bf16 per K step.
// ---------------------------------------------------------------------------
template <bool OUTBF, bool GELU, bool RES, bool TRANSOUT, bool BIAS>
__global__ __launch_bounds__(128) void gemm_wmma_bf16_kernel(
        const bf16* __restrict__ A, const bf16* __restrict__ Bt,
        const float* __restrict__ bias, const float* __restrict__ residual,
        void* __restrict__ outP, float alpha,
        int M, int N, int K, int lda, int ldb, int ldo) {
    (void)M;
    __shared__ __align__(16) bf16 As[2][BM * LDSS];
    __shared__ __align__(16) bf16 Bs[2][BN * LDSS];

    const int tid  = threadIdx.x;
    const int lane = tid & 31;
    const int wave = tid >> 5;
    const int wm   = wave >> 1;           // 0..1  (32-row strip)
    const int wn   = wave & 1;            // 0..1  (64-col strip)
    const int hh   = lane >> 4;           // lane half (K-half selector)
    const int l15  = lane & 15;

    const int blockM0 = blockIdx.y * BM;
    const int blockN0 = blockIdx.x * BN;

    v8f acc[2][4];
    const v8f vzero = {0.f, 0.f, 0.f, 0.f, 0.f, 0.f, 0.f, 0.f};
#pragma unroll
    for (int i = 0; i < 2; ++i)
#pragma unroll
        for (int j = 0; j < 4; ++j) acc[i][j] = vzero;

    const bf16* aTile = A  + (size_t)blockM0 * lda;   // + k0 per step
    const bf16* bTile = Bt + (size_t)blockN0 * ldb;
    const unsigned ldsA[2] = { (unsigned)(uintptr_t)&As[0][0], (unsigned)(uintptr_t)&As[1][0] };
    const unsigned ldsB[2] = { (unsigned)(uintptr_t)&Bs[0][0], (unsigned)(uintptr_t)&Bs[1][0] };

    const int nk = K / BK;
    if (wave == 0) {  // prologue: fetch tile 0
        tdm_load_tile_2d(ldsA[0], aTile, BM, (unsigned)K, (unsigned)M, (unsigned)lda);
        tdm_load_tile_2d(ldsB[0], bTile, BN, (unsigned)K, (unsigned)N, (unsigned)ldb);
    }

    int cur = 0;
    for (int kt = 0; kt < nk; ++kt) {
        if (wave == 0) {
            if (kt + 1 < nk) {   // prefetch next tile into alternate buffer (freed by last barrier)
                tdm_load_tile_2d(ldsA[cur ^ 1], aTile + (kt + 1) * BK, BM,
                                 (unsigned)K, (unsigned)M, (unsigned)lda);
                tdm_load_tile_2d(ldsB[cur ^ 1], bTile + (kt + 1) * BK, BN,
                                 (unsigned)K, (unsigned)N, (unsigned)ldb);
                __builtin_amdgcn_s_wait_tensorcnt((short)2);  // current tile's 2 ops done
            } else {
                __builtin_amdgcn_s_wait_tensorcnt((short)0);
            }
        }
        __syncthreads();          // current buffer visible to all waves

        FragU af[2], bfr[4];
#pragma unroll
        for (int i = 0; i < 2; ++i) {
            const int r = wm * 32 + i * 16 + l15;
            af[i].u[0] = *(const v4u*)(&As[cur][r * LDSS + 8 * hh]);
            af[i].u[1] = *(const v4u*)(&As[cur][r * LDSS + 16 + 8 * hh]);
        }
#pragma unroll
        for (int j = 0; j < 4; ++j) {
            const int c = wn * 64 + j * 16 + l15;
            bfr[j].u[0] = *(const v4u*)(&Bs[cur][c * LDSS + 8 * hh]);
            bfr[j].u[1] = *(const v4u*)(&Bs[cur][c * LDSS + 16 + 8 * hh]);
        }
#pragma unroll
        for (int i = 0; i < 2; ++i)
#pragma unroll
            for (int j = 0; j < 4; ++j)
                acc[i][j] = __builtin_amdgcn_wmma_f32_16x16x32_bf16(
                    false, af[i].v, false, bfr[j].v, (short)0, acc[i][j], false, false);

        __syncthreads();          // all waves done reading before buffer is refilled
        cur ^= 1;
    }

    // epilogue: C VGPR e maps to M = e + 8*half, N = lane&15 (ISA 16x16 f32 layout)
#pragma unroll
    for (int i = 0; i < 2; ++i) {
#pragma unroll
        for (int j = 0; j < 4; ++j) {
            const int colg = blockN0 + wn * 64 + j * 16 + l15;
#pragma unroll
            for (int e = 0; e < 8; ++e) {
                const int rowg = blockM0 + wm * 32 + i * 16 + e + 8 * hh;
                float v = acc[i][j][e] * alpha;
                if (BIAS) v += bias[colg];
                if (GELU) v = 0.5f * v * (1.f + erff(v * 0.70710678118f));
                if (RES)  v += residual[(size_t)rowg * ldo + colg];
                if (TRANSOUT) {
                    ((bf16*)outP)[(size_t)colg * ldo + rowg] = (bf16)v;
                } else if (OUTBF) {
                    ((bf16*)outP)[(size_t)rowg * ldo + colg] = (bf16)v;
                } else {
                    ((float*)outP)[(size_t)rowg * ldo + colg] = v;
                }
            }
        }
    }
}

// ---------------------------------------------------------------------------
// Host orchestration
// ---------------------------------------------------------------------------
extern "C" void kernel_launch(void* const* d_in, const int* in_sizes, int n_in,
                              void* d_out, int out_size, void* d_ws, size_t ws_size,
                              hipStream_t stream) {
    (void)in_sizes; (void)n_in; (void)out_size; (void)ws_size;
    const float* x     = (const float*)d_in[0];
    const float* wq    = (const float*)d_in[1];
    const float* bq    = (const float*)d_in[2];
    const float* wk    = (const float*)d_in[3];
    const float* bk    = (const float*)d_in[4];
    const float* wv    = (const float*)d_in[5];
    const float* bv    = (const float*)d_in[6];
    const float* wo    = (const float*)d_in[7];
    const float* bo    = (const float*)d_in[8];
    const float* w1    = (const float*)d_in[9];
    const float* b1    = (const float*)d_in[10];
    const float* w2    = (const float*)d_in[11];
    const float* b2    = (const float*)d_in[12];
    const float* g1    = (const float*)d_in[13];
    const float* bta1  = (const float*)d_in[14];
    const float* g2    = (const float*)d_in[15];
    const float* bta2  = (const float*)d_in[16];
    float* out = (float*)d_out;

    const int D = DIMD, H = DIMH, S = SEQ, M = MTOT;

    // workspace carve-up (deterministic each call)
    char* wsp = (char*)d_ws;
    auto carve = [&](size_t bytes) -> char* {
        char* p = wsp;
        wsp += (bytes + 255) & ~(size_t)255;
        return p;
    };
    bf16* hA   = (bf16*)carve((size_t)M * D * 2);   // ln1(x)
    bf16* qA   = (bf16*)carve((size_t)M * D * 2);
    bf16* kA   = (bf16*)carve((size_t)M * D * 2);
    bf16* vT   = (bf16*)carve((size_t)M * D * 2);   // [D][M]  (per-batch columns)
    bf16* ctx  = (bf16*)carve((size_t)M * D * 2);
    bf16* h2A  = (bf16*)carve((size_t)M * D * 2);   // ln2(x1)
    bf16* wqT  = (bf16*)carve((size_t)D * D * 2);
    bf16* wkT  = (bf16*)carve((size_t)D * D * 2);
    bf16* wvT  = (bf16*)carve((size_t)D * D * 2);
    bf16* woT  = (bf16*)carve((size_t)D * D * 2);
    bf16* w1T  = (bf16*)carve((size_t)D * H * 2);
    bf16* w2T  = (bf16*)carve((size_t)H * D * 2);
    bf16* G    = (bf16*)carve((size_t)M * H * 2);   // gelu(ffn1)
    float* scF = (float*)carve((size_t)S * S * 4);  // per-batch fp32 scores (reused)
    bf16* P    = (bf16*)carve((size_t)S * S * 2);   // per-batch bf16 probs (reused)

    const dim3 tb256(32, 8);
    // 1) pack weights (fp32 [K][N] -> bf16 [N][K])
    pack_wT_kernel<<<dim3(D / 32, D / 32), tb256, 0, stream>>>(wq, wqT, D, D);
    pack_wT_kernel<<<dim3(D / 32, D / 32), tb256, 0, stream>>>(wk, wkT, D, D);
    pack_wT_kernel<<<dim3(D / 32, D / 32), tb256, 0, stream>>>(wv, wvT, D, D);
    pack_wT_kernel<<<dim3(D / 32, D / 32), tb256, 0, stream>>>(wo, woT, D, D);
    pack_wT_kernel<<<dim3(H / 32, D / 32), tb256, 0, stream>>>(w1, w1T, D, H);
    pack_wT_kernel<<<dim3(D / 32, H / 32), tb256, 0, stream>>>(w2, w2T, H, D);

    // 2) h = LN1(x)  (bf16)
    layernorm_bf16_kernel<<<M, 256, 0, stream>>>(x, g1, bta1, hA, D);

    // 3) q, k (bf16, row-major) ; v stored transposed -> vT[D][M]
    gemm_wmma_bf16_kernel<true,  false, false, false, true>
        <<<dim3(D / BN, M / BM), 128, 0, stream>>>(hA, wqT, bq, nullptr, qA, 1.f, M, D, D, D, D, D);
    gemm_wmma_bf16_kernel<true,  false, false, false, true>
        <<<dim3(D / BN, M / BM), 128, 0, stream>>>(hA, wkT, bk, nullptr, kA, 1.f, M, D, D, D, D, D);
    gemm_wmma_bf16_kernel<true,  false, false, true,  true>
        <<<dim3(D / BN, M / BM), 128, 0, stream>>>(hA, wvT, bv, nullptr, vT, 1.f, M, D, D, D, D, M);

    // 4) attention per batch (stream-serialized; scF/P reused)
    const float scale = 0.03125f;  // 1/sqrt(1024)
    for (int b = 0; b < NB; ++b) {
        const bf16* qb = qA + (size_t)b * S * D;
        const bf16* kb = kA + (size_t)b * S * D;
        // scores = scale * q @ k^T   (Bt == k row-major: exactly [N][K])
        gemm_wmma_bf16_kernel<false, false, false, false, false>
            <<<dim3(S / BN, S / BM), 128, 0, stream>>>(qb, kb, nullptr, nullptr, scF, scale,
                                                       S, S, D, D, D, S);
        softmax_bf16_kernel<<<S, 256, 0, stream>>>(scF, P, S);
        // ctx_b = P @ v_b  with Bt = vT columns of batch b (ldb = M)
        gemm_wmma_bf16_kernel<true,  false, false, false, false>
            <<<dim3(D / BN, S / BM), 128, 0, stream>>>(P, vT + (size_t)b * S, nullptr, nullptr,
                                                       ctx + (size_t)b * S * D, 1.f,
                                                       S, D, S, S, M, D);
    }

    // 5) x1 = x + ctx @ wo^T + bo   (fp32 -> d_out)
    gemm_wmma_bf16_kernel<false, false, true, false, true>
        <<<dim3(D / BN, M / BM), 128, 0, stream>>>(ctx, woT, bo, x, out, 1.f, M, D, D, D, D, D);

    // 6) h2 = LN2(x1)
    layernorm_bf16_kernel<<<M, 256, 0, stream>>>(out, g2, bta2, h2A, D);

    // 7) G = gelu(h2 @ w1^T + b1)  (bf16)
    gemm_wmma_bf16_kernel<true,  true,  false, false, true>
        <<<dim3(H / BN, M / BM), 128, 0, stream>>>(h2A, w1T, b1, nullptr, G, 1.f, M, H, D, D, D, H);

    // 8) y = x1 + G @ w2^T + b2  (fp32 -> d_out, in-place residual)
    gemm_wmma_bf16_kernel<false, false, true, false, true>
        <<<dim3(D / BN, M / BM), 128, 0, stream>>>(G, w2T, b2, out, out, 1.f, M, D, H, H, H, D);
}
